// BitAttention_12094627905999
// MI455X (gfx1250) — matrix operand order repaired and checked by reading, not compiled, not run on hardware
//
#include <hip/hip_runtime.h>

#define DIMN 2048
#define ROWS 4096   // B*S = 2*2048
#define SEQ  2048
#define NHEAD 16
#define HDIM 128

typedef __attribute__((ext_vector_type(16))) _Float16 v16h;
typedef __attribute__((ext_vector_type(8)))  _Float16 v8h;
typedef __attribute__((ext_vector_type(8)))  float    v8f;
typedef __attribute__((ext_vector_type(4)))  int      v4i;

#if __has_builtin(__builtin_amdgcn_global_load_async_to_lds_b128) && \
    __has_builtin(__builtin_amdgcn_s_wait_asynccnt)
#define HAVE_ASYNC 1
#else
#define HAVE_ASYNC 0
#endif

#if HAVE_ASYNC
typedef __attribute__((address_space(1))) v4i gv4i;  // global int4
typedef __attribute__((address_space(3))) v4i lv4i;  // LDS int4
__device__ __forceinline__ void async_ld16(const _Float16* g, _Float16* l) {
  // 16 bytes global -> LDS, no VGPR staging, tracked by ASYNCcnt
  __builtin_amdgcn_global_load_async_to_lds_b128((gv4i*)g, (lv4i*)l, 0, 0);
}
__device__ __forceinline__ void async_wait0() {
  __builtin_amdgcn_s_wait_asynccnt(0);
}
#else
__device__ __forceinline__ void async_wait0() {}
#endif

__device__ __forceinline__ v8f wmma_f16(v16h a, v16h b, v8f c) {
  // 8 args: (neg_a, A, neg_b, B, c_mod, C, reuse_a, reuse_b)
  return __builtin_amdgcn_wmma_f32_16x16x32_f16(false, a, false, b, (short)0, c,
                                                false, false);
}

// ---- monotone float<->uint encoding for atomic min/max ----
__device__ __forceinline__ unsigned fenc(float f) {
  unsigned u = __float_as_uint(f);
  return (u >> 31) ? ~u : (u | 0x80000000u);
}
__device__ __forceinline__ float fdec(unsigned e) {
  return (e >> 31) ? __uint_as_float(e & 0x7FFFFFFFu) : __uint_as_float(~e);
}

// A-fragment (16x32 f16), row-major tile t[row*ldp + col].
// lane(0-15): row=lane, K = hi*8 + {0..7} then 16 + hi*8 + {0..7}
__device__ __forceinline__ v16h load_fragA(const _Float16* t, int rowBase,
                                           int colBase, int ldp) {
  const int lane = threadIdx.x & 31;
  const int r = lane & 15, hi = lane >> 4;
  const _Float16* p = t + (rowBase + r) * ldp + colBase + hi * 8;
  union { v16h v; v8h h[2]; } u;
  u.h[0] = *(const v8h*)(p);
  u.h[1] = *(const v8h*)(p + 16);
  return u.v;
}

// B-fragment (32x16 f16): lane holds column n = lane&15, K = hi*8 + {0..7} then 16 + hi*8 + {0..7}
// (table order). Tile stored row-major by n: t[n*ldp + k].
__device__ __forceinline__ v16h load_fragB(const _Float16* t, int rowBase,
                                           int colBase, int ldp) {
  const int lane = threadIdx.x & 31;
  const int r = lane & 15, hi = lane >> 4;
  const _Float16* p = t + (rowBase + r) * ldp + colBase + hi * 8;
  union { v16h v; v8h h[2]; } u;
  u.h[0] = *(const v8h*)(p);
  u.h[1] = *(const v8h*)(p + 16);
  return u.v;
}

// ---------------- small utility kernels ----------------
__global__ void ba_init_scalars(unsigned* sc) {
  const int t = threadIdx.x;
  if (t < 4) ((float*)sc)[t] = 0.0f;                   // abs-sums
  else if (t < 12) sc[t] = (t & 1) ? 0u : 0xFFFFFFFFu; // even=min, odd=max init
}

__global__ void ba_abssum(const float* __restrict__ w, const float* __restrict__ s,
                          float* acc, long n) {
  const float sv = fabsf(s[0]);
  float sum = 0.0f;
  for (long i = (long)blockIdx.x * blockDim.x + threadIdx.x; i < n;
       i += (long)gridDim.x * blockDim.x)
    sum += fabsf(w[i]);
  atomicAdd(acc, sum * sv);
}

__global__ void ba_ternarize(const float* __restrict__ w, const float* __restrict__ s,
                             const float* __restrict__ acc, _Float16* __restrict__ wh,
                             long n) {
  const float sv = s[0];
  const float thr = 0.7f * acc[0] / (float)n;  // 0.7 * mean(|w*s|)
  for (long i = (long)blockIdx.x * blockDim.x + threadIdx.x; i < n;
       i += (long)gridDim.x * blockDim.x) {
    const float v = w[i] * sv;
    wh[i] = (_Float16)((v > thr) ? 1.0f : ((v < -thr) ? -1.0f : 0.0f));
  }
}

__global__ void ba_f32_to_f16(const float* __restrict__ x, _Float16* __restrict__ y,
                              long n) {
  for (long i = (long)blockIdx.x * blockDim.x + threadIdx.x; i < n;
       i += (long)gridDim.x * blockDim.x)
    y[i] = (_Float16)x[i];
}

__global__ void ba_requant(_Float16* __restrict__ y, const unsigned* mn,
                           const unsigned* mx, float* __restrict__ out32, long n) {
  const float lo = fdec(*mn), hiv = fdec(*mx);
  const float scale = 255.0f / (hiv - lo);
  const float inv = (hiv - lo) / 255.0f;
  for (long i = (long)blockIdx.x * blockDim.x + threadIdx.x; i < n;
       i += (long)gridDim.x * blockDim.x) {
    float v = (float)y[i];
    v = rintf(v * scale) * inv;
    if (out32) out32[i] = v;
    else       y[i] = (_Float16)v;
  }
}

// ---------------- WMMA GEMM: Y(f16) = X(f16) * W^T(f16 ternary) + b ----------------
// X: ROWS x DIMN row-major, W: DIMN(out) x DIMN(in) row-major, Y: ROWS x DIMN.
// Double-buffered LDS, one barrier per 32-wide k-step; async global->LDS when
// the toolchain exposes the gfx1250 async builtins.
__global__ __launch_bounds__(256) void ba_gemm_bit(
    const _Float16* __restrict__ A, const _Float16* __restrict__ W,
    const float* __restrict__ bias, _Float16* __restrict__ Y,
    unsigned* mn, unsigned* mx) {
  __shared__ __align__(16) _Float16 sA[2][128 * 40];
  __shared__ __align__(16) _Float16 sB[2][128 * 40];
  const int t = threadIdx.x;
  const int wv = t >> 5;
  const int waveM = wv >> 1;      // 0..3 -> 32-row strip
  const int waveN = wv & 1;       // 0..1 -> 64-col strip
  const int lane = t & 31, r = lane & 15, hi = lane >> 4;
  const int lrow = t >> 1;        // 0..127
  const int lcol = (t & 1) << 4;  // 0 / 16
  const size_t m0 = (size_t)blockIdx.x * 128;
  const size_t n0 = (size_t)blockIdx.y * 128;

  v8f zero = {};
  v8f acc[2][4];
  for (int i = 0; i < 2; ++i)
    for (int j = 0; j < 4; ++j) acc[i][j] = zero;

  auto stage = [&](int buf, int k0) {
    _Float16* la = &sA[buf][0] + lrow * 40 + lcol;
    _Float16* lb = &sB[buf][0] + lrow * 40 + lcol;
    const _Float16* ga = A + (m0 + lrow) * DIMN + k0 + lcol;
    const _Float16* gw = W + (n0 + lrow) * DIMN + k0 + lcol;
#if HAVE_ASYNC
    async_ld16(ga, la);
    async_ld16(ga + 8, la + 8);
    async_ld16(gw, lb);
    async_ld16(gw + 8, lb + 8);
#else
    *(v8h*)(la)     = *(const v8h*)(ga);
    *(v8h*)(la + 8) = *(const v8h*)(ga + 8);
    *(v8h*)(lb)     = *(const v8h*)(gw);
    *(v8h*)(lb + 8) = *(const v8h*)(gw + 8);
#endif
  };

  stage(0, 0);
  async_wait0();
  __syncthreads();

  for (int k0 = 0; k0 < DIMN; k0 += 32) {
    const int cur = (k0 >> 5) & 1;
    if (k0 + 32 < DIMN) stage(cur ^ 1, k0 + 32);  // prefetch next tile

    const _Float16* cA = &sA[cur][0];
    const _Float16* cB = &sB[cur][0];
    const v16h a0 = load_fragA(cA, waveM * 32, 0, 40);
    const v16h a1 = load_fragA(cA, waveM * 32 + 16, 0, 40);
#pragma unroll
    for (int nf = 0; nf < 4; ++nf) {
      const v16h bf = load_fragB(cB, waveN * 64 + nf * 16, 0, 40);
      acc[0][nf] = wmma_f16(a0, bf, acc[0][nf]);
      acc[1][nf] = wmma_f16(a1, bf, acc[1][nf]);
    }
    async_wait0();
    __syncthreads();
  }

  float lmin = 3.0e38f, lmax = -3.0e38f;
#pragma unroll
  for (int nf = 0; nf < 4; ++nf) {
    const size_t gn = n0 + waveN * 64 + nf * 16 + r;
    const float bv = bias[gn];
#pragma unroll
    for (int mf = 0; mf < 2; ++mf) {
      const size_t gm = m0 + waveM * 32 + mf * 16 + hi * 8;
#pragma unroll
      for (int rr = 0; rr < 8; ++rr) {
        float v = acc[mf][nf][rr] + bv;
        lmin = fminf(lmin, v);
        lmax = fmaxf(lmax, v);
        Y[(gm + rr) * DIMN + gn] = (_Float16)v;
      }
    }
  }
  atomicMin(mn, fenc(lmin));
  atomicMax(mx, fenc(lmax));
}

// ---------------- Flash attention with WMMA ----------------
// Grid: (SEQ/64, NHEAD, B). Block: 128 threads = 4 waves, 16 queries each.
__global__ __launch_bounds__(128) void ba_attention(
    const _Float16* __restrict__ Q, const _Float16* __restrict__ K,
    const _Float16* __restrict__ V, _Float16* __restrict__ O) {
  __shared__ __align__(16) _Float16 sK[32 * 136];    // key-major:  sK[key][dim]
  __shared__ __align__(16) _Float16 sV[128 * 40];    // dim-major:  sV[dim][key]
  __shared__ __align__(16) _Float16 sP[4 * 16 * 40]; // per-wave P scratch
  const int t = threadIdx.x;
  const int w = t >> 5, lane = t & 31, r = lane & 15, hi = lane >> 4;
  const int qt = blockIdx.x, h = blockIdx.y, b = blockIdx.z;
  const size_t qrow0 = (size_t)b * SEQ + (size_t)qt * 64 + (size_t)w * 16;
  const _Float16* qbase = Q + qrow0 * DIMN + h * HDIM;

  v16h qf[4];
#pragma unroll
  for (int dc = 0; dc < 4; ++dc) qf[dc] = load_fragA(qbase, 0, dc * 32, DIMN);

  v8f vz = {};
  float m_[8], l_[8];
  v8f o_[8];
#pragma unroll
  for (int i = 0; i < 8; ++i) { m_[i] = -1.0e30f; l_[i] = 0.0f; o_[i] = vz; }

  const int crow = t >> 2;        // 0..31 (key within chunk)
  const int ccol = (t & 3) * 32;  // dim chunk base
  _Float16* Pw = sP + w * (16 * 40);

  for (int kc = 0; kc < SEQ; kc += 32) {
    const size_t krow = (size_t)b * SEQ + kc + crow;
    const _Float16* kg = K + krow * DIMN + h * HDIM + ccol;
    const _Float16* vg = V + krow * DIMN + h * HDIM + ccol;
#pragma unroll
    for (int j = 0; j < 4; ++j) {
#if HAVE_ASYNC
      async_ld16(kg + j * 8, sK + crow * 136 + ccol + j * 8);
#else
      *(v8h*)(sK + crow * 136 + ccol + j * 8) = *(const v8h*)(kg + j * 8);
#endif
      v8h vv = *(const v8h*)(vg + j * 8);
#pragma unroll
      for (int e = 0; e < 8; ++e)
        sV[(ccol + j * 8 + e) * 40 + crow] = vv[e];  // transpose into LDS
    }
    async_wait0();
    __syncthreads();

    // S = Q * K^T  (two 16x16 tiles: keys 0..15, 16..31)
    v8f s0 = vz, s1 = vz;
#pragma unroll
    for (int dc = 0; dc < 4; ++dc) {
      const v16h bk0 = load_fragB(sK, 0,  dc * 32, 136);
      const v16h bk1 = load_fragB(sK, 16, dc * 32, 136);
      s0 = wmma_f16(qf[dc], bk0, s0);
      s1 = wmma_f16(qf[dc], bk1, s1);
    }
    const float sc = 0.08838834764831845f; // 1/sqrt(128)

    // online softmax; C-frag row = rr + 8*hi, col = lane&15
#pragma unroll
    for (int rr = 0; rr < 8; ++rr) {
      const float a = s0[rr] * sc, bb = s1[rr] * sc;
      float mx2 = fmaxf(a, bb);
#pragma unroll
      for (int msk = 1; msk < 16; msk <<= 1)
        mx2 = fmaxf(mx2, __shfl_xor(mx2, msk, 32));
      const float mnew = fmaxf(m_[rr], mx2);
      const float alpha = __expf(m_[rr] - mnew);
      m_[rr] = mnew;
      const float p0 = __expf(a - mnew), p1 = __expf(bb - mnew);
      Pw[(rr + hi * 8) * 40 + r]      = (_Float16)p0;
      Pw[(rr + hi * 8) * 40 + 16 + r] = (_Float16)p1;
      float ps = p0 + p1;
#pragma unroll
      for (int msk = 1; msk < 16; msk <<= 1)
        ps += __shfl_xor(ps, msk, 32);
      l_[rr] = l_[rr] * alpha + ps;
#pragma unroll
      for (int f = 0; f < 8; ++f) o_[f][rr] *= alpha;
    }

    // O += P * V  (P: 16x32 as A-frag; V^T tiles as B-frags)
    const v16h pa = load_fragA(Pw, 0, 0, 40);
#pragma unroll
    for (int f = 0; f < 8; ++f) {
      const v16h bv2 = load_fragB(sV, f * 16, 0, 40);
      o_[f] = wmma_f16(pa, bv2, o_[f]);
    }
    __syncthreads();
  }

#pragma unroll
  for (int f = 0; f < 8; ++f) {
    const size_t dcol = (size_t)h * HDIM + f * 16 + r;
#pragma unroll
    for (int rr = 0; rr < 8; ++rr) {
      const float vo = o_[f][rr] / l_[rr];
      O[(qrow0 + rr + hi * 8) * DIMN + dcol] = (_Float16)vo;
    }
  }
}

// ---------------- launcher ----------------
extern "C" void kernel_launch(void* const* d_in, const int* in_sizes, int n_in,
                              void* d_out, int out_size, void* d_ws, size_t ws_size,
                              hipStream_t stream) {
  (void)in_sizes; (void)n_in; (void)out_size; (void)ws_size;
  const float* x = (const float*)d_in[0];
  const float* w_[4] = {(const float*)d_in[1], (const float*)d_in[4],
                        (const float*)d_in[7], (const float*)d_in[10]};
  const float* b_[4] = {(const float*)d_in[2], (const float*)d_in[5],
                        (const float*)d_in[8], (const float*)d_in[11]};
  const float* s_[4] = {(const float*)d_in[3], (const float*)d_in[6],
                        (const float*)d_in[9], (const float*)d_in[12]};

  char* ws = (char*)d_ws;
  unsigned* sc = (unsigned*)ws;  // [0..3] abssum f32; min[i]=4+2i, max[i]=5+2i
  size_t off = 256;
  _Float16* xh = (_Float16*)(ws + off); off += (size_t)ROWS * DIMN * 2;
  _Float16* wh[4];
  for (int i = 0; i < 4; ++i) {
    wh[i] = (_Float16*)(ws + off); off += (size_t)DIMN * DIMN * 2;
  }
  _Float16* qh = (_Float16*)(ws + off); off += (size_t)ROWS * DIMN * 2;
  _Float16* kh = (_Float16*)(ws + off); off += (size_t)ROWS * DIMN * 2;
  _Float16* vh = (_Float16*)(ws + off); off += (size_t)ROWS * DIMN * 2;
  _Float16* ah = (_Float16*)(ws + off); off += (size_t)ROWS * DIMN * 2;
  _Float16* yh = xh;  // reuse: x no longer needed after q/k/v GEMMs

  const long NW = (long)DIMN * DIMN;  // 4M weight elements
  const long NA = (long)ROWS * DIMN;  // 8M activation elements

  ba_init_scalars<<<1, 64, 0, stream>>>(sc);
  for (int i = 0; i < 4; ++i)
    ba_abssum<<<512, 256, 0, stream>>>(w_[i], s_[i], (float*)sc + i, NW);
  for (int i = 0; i < 4; ++i)
    ba_ternarize<<<2048, 256, 0, stream>>>(w_[i], s_[i], (const float*)sc + i,
                                           wh[i], NW);
  ba_f32_to_f16<<<2048, 256, 0, stream>>>(x, xh, NA);

  dim3 gg(ROWS / 128, DIMN / 128);
  ba_gemm_bit<<<gg, 256, 0, stream>>>(xh, wh[0], b_[0], qh, sc + 4, sc + 5);
  ba_gemm_bit<<<gg, 256, 0, stream>>>(xh, wh[1], b_[1], kh, sc + 6, sc + 7);
  ba_gemm_bit<<<gg, 256, 0, stream>>>(xh, wh[2], b_[2], vh, sc + 8, sc + 9);

  ba_requant<<<2048, 256, 0, stream>>>(qh, sc + 4, sc + 5, nullptr, NA);
  ba_requant<<<2048, 256, 0, stream>>>(kh, sc + 6, sc + 7, nullptr, NA);
  ba_requant<<<2048, 256, 0, stream>>>(vh, sc + 8, sc + 9, nullptr, NA);

  ba_attention<<<dim3(SEQ / 64, NHEAD, 2), 128, 0, stream>>>(qh, kh, vh, ah);

  ba_gemm_bit<<<gg, 256, 0, stream>>>(ah, wh[3], b_[3], yh, sc + 10, sc + 11);
  ba_requant<<<2048, 256, 0, stream>>>(yh, sc + 10, sc + 11, (float*)d_out, NA);
}
